// FeedForwardQuantum_65481071404432
// MI455X (gfx1250) — matrix operand order repaired and checked by reading, not compiled, not run on hardware
//
#include <hip/hip_runtime.h>

// ---------------------------------------------------------------------------
// Types for CDNA5 WMMA (wave32)
// ---------------------------------------------------------------------------
typedef __bf16 bf16_t;
typedef __attribute__((ext_vector_type(16))) __bf16 v16bf;
typedef __attribute__((ext_vector_type(8)))  float  v8f;

union V16Frag {
    v16bf v;
    uint4 q[2];
};

// ---------------------------------------------------------------------------
// Problem dims (compile-time; reference is fixed-shape)
// ---------------------------------------------------------------------------
#define EMBED 1024
#define FFN   4096
#define MTOK  (4 * 4096)   // BATCH * SEQ = 16384 rows

// ---------------------------------------------------------------------------
// CDNA5 async global->LDS copy (ASYNCcnt-tracked, bypasses VGPR staging).
// LDS destination VGPR holds the wave-relative LDS byte offset = low 32 bits
// of the flat address (see ISA 10.2, LDS aperture).
// ---------------------------------------------------------------------------
__device__ __forceinline__ unsigned lds_off(const void* p) {
    return (unsigned)(unsigned long long)p;
}

__device__ __forceinline__ void async_ld_b128(unsigned lds_byte,
                                              const void* gptr) {
    asm volatile("global_load_async_to_lds_b128 %0, %1, off"
                 :: "v"(lds_byte), "v"((unsigned long long)gptr)
                 : "memory");
}

// ---------------------------------------------------------------------------
// Kernel 1: q = cos(x + theta) -> bf16   (8 elements / thread, b128 traffic)
// ---------------------------------------------------------------------------
__global__ __launch_bounds__(256) void qprep_kernel(const float* __restrict__ x,
                                                    const float* __restrict__ theta,
                                                    bf16_t* __restrict__ q) {
    const long long i = ((long long)blockIdx.x * blockDim.x + threadIdx.x) * 8;
    const int col = (int)(i & (EMBED - 1));

    float4 a0 = *(const float4*)(x + i);
    float4 a1 = *(const float4*)(x + i + 4);
    float4 t0 = *(const float4*)(theta + col);
    float4 t1 = *(const float4*)(theta + col + 4);

    float f[8] = { a0.x + t0.x, a0.y + t0.y, a0.z + t0.z, a0.w + t0.w,
                   a1.x + t1.x, a1.y + t1.y, a1.z + t1.z, a1.w + t1.w };

    struct alignas(16) BF8 { bf16_t v[8]; } o;
#pragma unroll
    for (int j = 0; j < 8; ++j) o.v[j] = (bf16_t)__cosf(f[j]);
    *(BF8*)(q + i) = o;
}

// ---------------------------------------------------------------------------
// Kernel 2: f32 -> bf16 convert (weights), 8 elements / thread
// ---------------------------------------------------------------------------
__global__ __launch_bounds__(256) void cvt_kernel(const float* __restrict__ src,
                                                  bf16_t* __restrict__ dst) {
    const long long i = ((long long)blockIdx.x * blockDim.x + threadIdx.x) * 8;
    float4 a0 = *(const float4*)(src + i);
    float4 a1 = *(const float4*)(src + i + 4);
    float f[8] = { a0.x, a0.y, a0.z, a0.w, a1.x, a1.y, a1.z, a1.w };
    struct alignas(16) BF8 { bf16_t v[8]; } o;
#pragma unroll
    for (int j = 0; j < 8; ++j) o.v[j] = (bf16_t)f[j];
    *(BF8*)(dst + i) = o;
}

// ---------------------------------------------------------------------------
// Kernel 3: bf16 WMMA GEMM:  C[M,N] = A[M,K] * Bw[N,K]^T   (fixed N, K)
//
// WG = 256 threads = 8 waves arranged 2(M) x 4(N).
// WG tile 128 x 256, K-step 32. Wave tile 64 x 64 = 4x4 WMMA accumulators.
// Double-buffered LDS tiles filled by GLOBAL_LOAD_ASYNC_TO_LDS_B128:
// 6 async b128 per thread per tile (2 A + 4 B). Steady state:
//   issue next tile -> s_wait_asynccnt 6 -> barrier -> 16 WMMA -> barrier
// Last K-step peeled so the hot loop is branch-free; unroll-by-2 makes the
// double-buffer index a compile-time constant (no cndmask address selects).
// ---------------------------------------------------------------------------
#define BM 128
#define BN 256
#define BK 32
#define LDS_STRIDE 40   // halves per row (32 data + 8 pad -> 80B row pitch)

template <bool RELU_BF16_OUT, int N, int K>
__global__ __launch_bounds__(256) void wmma_gemm_kernel(const bf16_t* __restrict__ A,
                                                        const bf16_t* __restrict__ Bw,
                                                        bf16_t* __restrict__ Cbf,
                                                        float* __restrict__ Cf) {
    __shared__ bf16_t As[2][BM * LDS_STRIDE];   // 2 x 10 KB
    __shared__ bf16_t Bs[2][BN * LDS_STRIDE];   // 2 x 20 KB

    const int tid  = threadIdx.x;
    const int lane = tid & 31;
    const int wave = tid >> 5;
    const int l16  = lane & 15;
    const int hi   = lane >> 4;          // 0: lanes 0-15, 1: lanes 16-31
    const int wm   = wave & 1;           // 64-row slab
    const int wn   = wave >> 1;          // 64-col slab

    const int m0 = blockIdx.y * BM;
    const int n0 = blockIdx.x * BN;

    // --- cooperative async tile-load assignments -------------------------
    const int arow = tid >> 1;                     // A: 2 chunks / thread
    const int acol = (tid & 1) * 16;               // half offset 0 or 16
    const bf16_t* ag = A + (size_t)(m0 + arow) * K + acol;
    const int brow = tid;                          // B: 1 row (4 chunks) / thread
    const bf16_t* bg = Bw + (size_t)(n0 + brow) * K;

    unsigned a_lds[2], b_lds[2];
#pragma unroll
    for (int b = 0; b < 2; ++b) {
        a_lds[b] = lds_off(&As[b][arow * LDS_STRIDE + acol]);
        b_lds[b] = lds_off(&Bs[b][brow * LDS_STRIDE]);
    }

    v8f acc[4][4] = {};

    constexpr int NK = K / BK;

    auto issue_tile = [&](int buf, int kk) {
        async_ld_b128(a_lds[buf],      ag + kk);
        async_ld_b128(a_lds[buf] + 16, ag + kk + 8);
#pragma unroll
        for (int c = 0; c < 4; ++c)
            async_ld_b128(b_lds[buf] + c * 16, bg + kk + c * 8);
    };

    auto mma_tile = [&](int cur) {
        V16Frag af[4], bfr[4];
#pragma unroll
        for (int i = 0; i < 4; ++i) {
            const int r  = wm * 64 + i * 16 + l16;   // A row within tile
            const int kb = hi * 8;                   // lane-half K phase
            af[i].q[0] = *(const uint4*)(&As[cur][r * LDS_STRIDE + kb]);
            af[i].q[1] = *(const uint4*)(&As[cur][r * LDS_STRIDE + kb + 16]);
        }
#pragma unroll
        for (int j = 0; j < 4; ++j) {
            const int c = wn * 64 + j * 16 + l16;    // B column (= Bw row)
            bfr[j].q[0] = *(const uint4*)(&Bs[cur][c * LDS_STRIDE + hi * 8]);
            bfr[j].q[1] = *(const uint4*)(&Bs[cur][c * LDS_STRIDE + hi * 8 + 16]);
        }
#pragma unroll
        for (int i = 0; i < 4; ++i)
#pragma unroll
            for (int j = 0; j < 4; ++j)
                acc[i][j] = __builtin_amdgcn_wmma_f32_16x16x32_bf16(
                    false, af[i].v, false, bfr[j].v,
                    (short)0, acc[i][j], false, false);
    };

    // prologue: fill buffer 0 for k-step 0
    issue_tile(0, 0);

    // steady state: branch-free, buffer index constant per unrolled copy
#pragma unroll 2
    for (int k0 = 0; k0 < NK - 1; ++k0) {
        const int cur = k0 & 1;
        issue_tile(cur ^ 1, (k0 + 1) * BK);
        asm volatile("s_wait_asynccnt 0x6" ::: "memory");
        __syncthreads();
        mma_tile(cur);
        __syncthreads();
    }

    // peeled final K-step
    asm volatile("s_wait_asynccnt 0x0" ::: "memory");
    __syncthreads();
    mma_tile((NK - 1) & 1);

    // --- epilogue: C/D 16x16 f32 layout -> memory ------------------------
#pragma unroll
    for (int i = 0; i < 4; ++i) {
#pragma unroll
        for (int j = 0; j < 4; ++j) {
            const int col = n0 + wn * 64 + j * 16 + l16;
#pragma unroll
            for (int r = 0; r < 8; ++r) {
                const int row = m0 + wm * 64 + i * 16 + hi * 8 + r;
                const float v = acc[i][j][r];
                if (RELU_BF16_OUT) {
                    Cbf[(size_t)row * N + col] = (bf16_t)fmaxf(v, 0.0f);
                } else {
                    Cf[(size_t)row * N + col] = v;
                }
            }
        }
    }
}

// ---------------------------------------------------------------------------
// Host-side launcher
// ---------------------------------------------------------------------------
extern "C" void kernel_launch(void* const* d_in, const int* in_sizes, int n_in,
                              void* d_out, int out_size, void* d_ws, size_t ws_size,
                              hipStream_t stream) {
    const float* x     = (const float*)d_in[0];   // [4, 4096, 1024]
    const float* theta = (const float*)d_in[1];   // [1024]
    const float* W1    = (const float*)d_in[2];   // [4096, 1024]
    const float* W2    = (const float*)d_in[3];   // [1024, 4096]
    float* out = (float*)d_out;                   // [4, 4096, 1024]

    char* ws = (char*)d_ws;
    bf16_t* qbf = (bf16_t*)(ws);                                   // 32 MB
    bf16_t* w1b = (bf16_t*)(ws + (size_t)MTOK * EMBED * 2);        //  8 MB
    bf16_t* w2b = (bf16_t*)(ws + (size_t)MTOK * EMBED * 2
                                + (size_t)FFN * EMBED * 2);        //  8 MB
    bf16_t* hbf = (bf16_t*)(ws + (size_t)MTOK * EMBED * 2
                                + (size_t)FFN * EMBED * 2 * 2);    // 128 MB

    qprep_kernel<<<(MTOK * (long long)EMBED) / (8 * 256), 256, 0, stream>>>(x, theta, qbf);
    cvt_kernel<<<((long long)FFN * EMBED) / (8 * 256), 256, 0, stream>>>(W1, w1b);
    cvt_kernel<<<((long long)EMBED * FFN) / (8 * 256), 256, 0, stream>>>(W2, w2b);

    // GEMM1 + ReLU: h[M, FFN] = relu(q * W1^T), bf16 out
    {
        dim3 grid(FFN / BN, MTOK / BM);   // (16, 128)
        wmma_gemm_kernel<true, FFN, EMBED><<<grid, 256, 0, stream>>>(
            qbf, w1b, hbf, nullptr);
    }
    // GEMM2: out[M, EMBED] = h * W2^T, f32 out
    {
        dim3 grid(EMBED / BN, MTOK / BM); // (4, 128)
        wmma_gemm_kernel<false, EMBED, FFN><<<grid, 256, 0, stream>>>(
            hbf, w2b, nullptr, out);
    }
}
